// MultiHeadedAttention_19928648253890
// MI455X (gfx1250) — matrix operand order repaired and checked by reading, not compiled, not run on hardware
//
#include <hip/hip_runtime.h>
#include <hip/hip_bf16.h>

// ---------------------------------------------------------------------------
// MultiHeadedAttention for MI455X (gfx1250, wave32, WMMA)
//   q = X Wq^T + b ; k,v similar ; concat KV cache ; softmax(QK^T/8) V ; Wo
// Strategy: bf16 WMMA (v_wmma_f32_16x16x32_bf16) everywhere, fp32 accumulate,
// fp32 softmax. Layouts chosen so every WMMA fragment load is contiguous.
// ---------------------------------------------------------------------------

typedef __bf16 bf16_t;
typedef __attribute__((ext_vector_type(16))) __bf16 v16bf;
typedef __attribute__((ext_vector_type(8)))  __bf16 v8bf;
typedef __attribute__((ext_vector_type(8)))  float  v8f;

#define WMMA_BF16(A, B, C) \
  __builtin_amdgcn_wmma_f32_16x16x32_bf16(false, (A), false, (B), (short)0, (C), false, false)

// Problem constants
constexpr int BATCH = 4;
constexpr int LQ    = 1024;
constexpr int DMODEL= 1024;
constexpr int HEADS = 16;
constexpr int DHEAD = 64;
constexpr int KVLEN = 2048;         // 1024 cache + 1024 new
constexpr int MROWS = BATCH * LQ;   // 4096

// ---------------------------------------------------------------------------
// fp32 -> bf16 elementwise convert
// ---------------------------------------------------------------------------
__global__ void f32_to_bf16_kernel(const float* __restrict__ src,
                                   bf16_t* __restrict__ dst, int n) {
  int i = blockIdx.x * blockDim.x + threadIdx.x;
  if (i < n) dst[i] = (bf16_t)src[i];
}

// ---------------------------------------------------------------------------
// Old cache ingest: (B,H,1024,128) fp32 ->
//   * new_cache output rows [0,1024)
//   * Kbuf  bf16 (B*H, 2048, 64)      rows [0,1024)
//   * Vtbuf bf16 (B*H, 64, 2048) (transposed) cols [0,1024)
// ---------------------------------------------------------------------------
__global__ void cache_ingest_kernel(const float* __restrict__ cache,
                                    float* __restrict__ cacheOut,
                                    bf16_t* __restrict__ Kb,
                                    bf16_t* __restrict__ Vt) {
  int i = blockIdx.x * blockDim.x + threadIdx.x;  // 0 .. B*H*1024*128
  int c  = i & 127;
  int s  = (i >> 7) & 1023;
  int bh = i >> 17;
  float v = cache[i];
  cacheOut[(((size_t)bh * KVLEN) + s) * 128 + c] = v;
  if (c < DHEAD)
    Kb[(((size_t)bh * KVLEN) + s) * DHEAD + c] = (bf16_t)v;
  else
    Vt[((size_t)bh * DHEAD + (c - DHEAD)) * KVLEN + s] = (bf16_t)v;
}

// ---------------------------------------------------------------------------
// GEMM: C[m][n] = sum_k X[m][k] * W[n][k] + bias[n]   (M=4096,N=1024,K=1024)
// Per-wave tile 32x64 (2 m-frags x 4 n-frags, 8 WMMAs per k-step of 32).
// A-frag: lane(row=lane&15) loads 2x16B contiguous (K split 16/16 per half-wave)
// B-frag: lane(col=lane&15) loads 32B contiguous from W row (K split 16/16)
// Epilogue scatters per MODE.
// ---------------------------------------------------------------------------
enum { MODE_Q = 0, MODE_K = 1, MODE_V = 2, MODE_O = 3 };

template <int MODE>
__global__ __launch_bounds__(256)
void proj_gemm_kernel(const bf16_t* __restrict__ X,    // (4096,1024) bf16
                      const bf16_t* __restrict__ W,    // (1024,1024) bf16
                      const float*  __restrict__ bias, // (1024) fp32
                      bf16_t* __restrict__ outB,       // mode-dependent bf16 buf
                      float*  __restrict__ outF)       // cache region / final out
{
  const int wid  = threadIdx.x >> 5;
  const int lane = threadIdx.x & 31;
  const int lm = lane & 15, hi = lane >> 4;
  const int gw = blockIdx.x * 8 + wid;   // 2048 waves total
  const int tileN = gw & 15;             // 16 tiles of 64 cols
  const int tileM = gw >> 4;             // 128 tiles of 32 rows
  const int m0 = tileM * 32, n0 = tileN * 64;

  v8f acc[2][4] = {};
  const bf16_t* aRow0 = X + (size_t)(m0 + lm) * DMODEL;
  const bf16_t* aRow1 = X + (size_t)(m0 + 16 + lm) * DMODEL;

  for (int k0 = 0; k0 < DMODEL; k0 += 32) {
    v16bf a[2];
    {
      v8bf lo0 = *(const v8bf*)(aRow0 + k0 + hi * 16);
      v8bf h0  = *(const v8bf*)(aRow0 + k0 + 8 + hi * 16);
      v8bf lo1 = *(const v8bf*)(aRow1 + k0 + hi * 16);
      v8bf h1  = *(const v8bf*)(aRow1 + k0 + 8 + hi * 16);
#pragma unroll
      for (int e = 0; e < 8; ++e) {
        a[0][e] = lo0[e]; a[0][8 + e] = h0[e];
        a[1][e] = lo1[e]; a[1][8 + e] = h1[e];
      }
    }
    if (k0 + 64 < DMODEL) {  // warm next A rows (weights stay resident in L2)
      __builtin_prefetch(aRow0 + k0 + 64, 0, 0);
      __builtin_prefetch(aRow1 + k0 + 64, 0, 0);
    }
#pragma unroll
    for (int nf = 0; nf < 4; ++nf) {
      const bf16_t* bp = W + (size_t)(n0 + nf * 16 + lm) * DMODEL + k0 + 16 * hi;
      v16bf bfr = *(const v16bf*)bp;  // 32B contiguous
      acc[0][nf] = WMMA_BF16(a[0], bfr, acc[0][nf]);
      acc[1][nf] = WMMA_BF16(a[1], bfr, acc[1][nf]);
    }
  }

  // Epilogue: C/D layout -> lane holds col n=lane&15, rows r + 8*(lane>>4)
#pragma unroll
  for (int nf = 0; nf < 4; ++nf) {
    const int n = n0 + nf * 16 + lm;
    const float bn = bias[n];
    const int h = n >> 6, dh = n & 63;
#pragma unroll
    for (int am = 0; am < 2; ++am) {
#pragma unroll
      for (int r = 0; r < 8; ++r) {
        const int gm = m0 + am * 16 + r + 8 * hi;
        const int b = gm >> 10, l = gm & 1023;
        const float v = acc[am][nf][r] + bn;
        if (MODE == MODE_Q) {
          outB[(((size_t)(b * HEADS + h) * LQ) + l) * DHEAD + dh] = (bf16_t)v;
        } else if (MODE == MODE_K) {
          const size_t s = (size_t)(b * HEADS + h) * KVLEN + 1024 + l;
          outB[s * DHEAD + dh] = (bf16_t)v;
          outF[s * 128 + dh] = v;
        } else if (MODE == MODE_V) {
          const size_t s = (size_t)(b * HEADS + h) * KVLEN + 1024 + l;
          outB[((size_t)(b * HEADS + h) * DHEAD + dh) * KVLEN + 1024 + l] = (bf16_t)v;
          outF[s * 128 + DHEAD + dh] = v;
        } else {  // MODE_O : final fp32 output
          outF[(size_t)gm * DMODEL + n] = v;
        }
      }
    }
  }
}

// ---------------------------------------------------------------------------
// Flash-style attention. One wave per 16-row Q tile of one (b,h).
// Chunks of 32 keys: S(16x32)=Q*K^T via 4 WMMAs, fp32 online softmax
// (half-wave __shfl_xor reductions), P -> LDS (layout swap) -> A-frag,
// O += P*V via 4 WMMAs against transposed V.
// ---------------------------------------------------------------------------
__global__ __launch_bounds__(256)
void attention_kernel(const bf16_t* __restrict__ Qb,  // (B*H,1024,64)
                      const bf16_t* __restrict__ Kb,  // (B*H,2048,64)
                      const bf16_t* __restrict__ Vt,  // (B*H,64,2048)
                      bf16_t* __restrict__ O)         // (4096,1024) merged heads
{
  __shared__ __align__(32) bf16_t pTile[8][16 * 32];  // 1KB per wave
  const int wid  = threadIdx.x >> 5;
  const int lane = threadIdx.x & 31;
  const int lm = lane & 15, hi = lane >> 4;
  const int gw = blockIdx.x * 8 + wid;  // 4096 waves
  const int bh = gw >> 6;               // (b*16+h)
  const int q0 = (gw & 63) * 16;

  const bf16_t* Qh = Qb + (size_t)bh * LQ * DHEAD;
  const bf16_t* Kh = Kb + (size_t)bh * KVLEN * DHEAD;
  const bf16_t* Vh = Vt + (size_t)bh * DHEAD * KVLEN;

  // Q A-fragments for k=[0,32) and [32,64)
  v16bf qa[2];
  {
    const bf16_t* qr = Qh + (size_t)(q0 + lm) * DHEAD;
    v8bf l0 = *(const v8bf*)(qr + hi * 16);
    v8bf h0 = *(const v8bf*)(qr + 8 + hi * 16);
    v8bf l1 = *(const v8bf*)(qr + 32 + hi * 16);
    v8bf h1 = *(const v8bf*)(qr + 40 + hi * 16);
#pragma unroll
    for (int e = 0; e < 8; ++e) {
      qa[0][e] = l0[e]; qa[0][8 + e] = h0[e];
      qa[1][e] = l1[e]; qa[1][8 + e] = h1[e];
    }
  }

  float rowMax[8], rowSum[8];
#pragma unroll
  for (int r = 0; r < 8; ++r) { rowMax[r] = -1e30f; rowSum[r] = 0.0f; }
  v8f o[4] = {};

  bf16_t* myP = pTile[wid];

  for (int j = 0; j < KVLEN; j += 32) {
    // ---- scores S0 (keys j..j+15), S1 (keys j+16..j+31)
    v8f s0 = {}, s1 = {};
    {
      const bf16_t* kr0 = Kh + (size_t)(j + lm) * DHEAD + 16 * hi;
      const bf16_t* kr1 = Kh + (size_t)(j + 16 + lm) * DHEAD + 16 * hi;
      v16bf b00 = *(const v16bf*)(kr0);
      v16bf b01 = *(const v16bf*)(kr0 + 32);
      v16bf b10 = *(const v16bf*)(kr1);
      v16bf b11 = *(const v16bf*)(kr1 + 32);
      s0 = WMMA_BF16(qa[0], b00, s0);
      s0 = WMMA_BF16(qa[1], b01, s0);
      s1 = WMMA_BF16(qa[0], b10, s1);
      s1 = WMMA_BF16(qa[1], b11, s1);
    }
    // ---- online softmax (scale 1/sqrt(64)=0.125 folded here, fp32)
    __builtin_amdgcn_wave_barrier();
#pragma unroll
    for (int r = 0; r < 8; ++r) {
      float x0 = s0[r] * 0.125f, x1 = s1[r] * 0.125f;
      float cm = fmaxf(x0, x1);
      cm = fmaxf(cm, __shfl_xor(cm, 1, 32));
      cm = fmaxf(cm, __shfl_xor(cm, 2, 32));
      cm = fmaxf(cm, __shfl_xor(cm, 4, 32));
      cm = fmaxf(cm, __shfl_xor(cm, 8, 32));
      const float nm = fmaxf(rowMax[r], cm);
      const float sc = __expf(rowMax[r] - nm);
      rowMax[r] = nm;
      const float p0 = __expf(x0 - nm);
      const float p1 = __expf(x1 - nm);
      float ps = p0 + p1;
      ps += __shfl_xor(ps, 1, 32);
      ps += __shfl_xor(ps, 2, 32);
      ps += __shfl_xor(ps, 4, 32);
      ps += __shfl_xor(ps, 8, 32);
      rowSum[r] = rowSum[r] * sc + ps;
#pragma unroll
      for (int f = 0; f < 4; ++f) o[f][r] *= sc;
      // C-layout -> row-major 16x32 bf16 tile in LDS
      myP[(r + 8 * hi) * 32 + lm]      = (bf16_t)p0;
      myP[(r + 8 * hi) * 32 + 16 + lm] = (bf16_t)p1;
    }
    __builtin_amdgcn_wave_barrier();   // per-wave LDS is in-order; fence compiler
    // ---- P as A-fragment (16x32)
    v16bf pa;
    {
      v8bf pl = *(const v8bf*)(myP + lm * 32 + hi * 16);
      v8bf ph = *(const v8bf*)(myP + lm * 32 + 8 + hi * 16);
#pragma unroll
      for (int e = 0; e < 8; ++e) { pa[e] = pl[e]; pa[8 + e] = ph[e]; }
    }
    __builtin_amdgcn_wave_barrier();
    // ---- O += P * V  (V transposed: contiguous 32B B-fragment loads)
#pragma unroll
    for (int f = 0; f < 4; ++f) {
      const bf16_t* vr = Vh + (size_t)(f * 16 + lm) * KVLEN + j + 16 * hi;
      v16bf vb = *(const v16bf*)vr;
      o[f] = WMMA_BF16(pa, vb, o[f]);
    }
  }

  // ---- normalize + write merged-head bf16 output (4096 x 1024)
  const int b = bh >> 4, h = bh & 15;
#pragma unroll
  for (int r = 0; r < 8; ++r) {
    const float inv = 1.0f / rowSum[r];
    const int gm = b * LQ + q0 + r + 8 * hi;
#pragma unroll
    for (int f = 0; f < 4; ++f) {
      O[(size_t)gm * DMODEL + h * DHEAD + f * 16 + lm] = (bf16_t)(o[f][r] * inv);
    }
  }
}

// ---------------------------------------------------------------------------
// Host-side launcher
// ---------------------------------------------------------------------------
extern "C" void kernel_launch(void* const* d_in, const int* in_sizes, int n_in,
                              void* d_out, int out_size, void* d_ws, size_t ws_size,
                              hipStream_t stream) {
  const float* query = (const float*)d_in[0];
  const float* key   = (const float*)d_in[1];
  const float* value = (const float*)d_in[2];
  const float* cache = (const float*)d_in[3];
  const float* Wq = (const float*)d_in[4];
  const float* bq = (const float*)d_in[5];
  const float* Wk = (const float*)d_in[6];
  const float* bk = (const float*)d_in[7];
  const float* Wv = (const float*)d_in[8];
  const float* bv = (const float*)d_in[9];
  const float* Wo = (const float*)d_in[10];
  const float* bo = (const float*)d_in[11];

  float* out      = (float*)d_out;                         // (B,L,D)
  float* cacheOut = (float*)d_out + (size_t)MROWS * DMODEL;// (B,H,2048,128)

  // Workspace carve-up (bytes)
  char* ws = (char*)d_ws;
  size_t off = 0;
  auto carve = [&](size_t bytes) { char* p = ws + off; off += (bytes + 255) & ~size_t(255); return p; };
  bf16_t* Xq  = (bf16_t*)carve((size_t)MROWS * DMODEL * 2);
  bf16_t* Xk  = (bf16_t*)carve((size_t)MROWS * DMODEL * 2);
  bf16_t* Xv  = (bf16_t*)carve((size_t)MROWS * DMODEL * 2);
  bf16_t* Wqb = (bf16_t*)carve((size_t)DMODEL * DMODEL * 2);
  bf16_t* Wkb = (bf16_t*)carve((size_t)DMODEL * DMODEL * 2);
  bf16_t* Wvb = (bf16_t*)carve((size_t)DMODEL * DMODEL * 2);
  bf16_t* Wob = (bf16_t*)carve((size_t)DMODEL * DMODEL * 2);
  bf16_t* Qb  = (bf16_t*)carve((size_t)BATCH * HEADS * LQ * DHEAD * 2);
  bf16_t* Kb  = (bf16_t*)carve((size_t)BATCH * HEADS * KVLEN * DHEAD * 2);
  bf16_t* Vt  = (bf16_t*)carve((size_t)BATCH * HEADS * DHEAD * KVLEN * 2);
  bf16_t* Ao  = (bf16_t*)carve((size_t)MROWS * DMODEL * 2);

  const int TPB = 256;
  const int nAct = MROWS * DMODEL;    // 4,194,304
  const int nW   = DMODEL * DMODEL;   // 1,048,576
  f32_to_bf16_kernel<<<(nAct + TPB - 1) / TPB, TPB, 0, stream>>>(query, Xq, nAct);
  f32_to_bf16_kernel<<<(nAct + TPB - 1) / TPB, TPB, 0, stream>>>(key,   Xk, nAct);
  f32_to_bf16_kernel<<<(nAct + TPB - 1) / TPB, TPB, 0, stream>>>(value, Xv, nAct);
  f32_to_bf16_kernel<<<(nW + TPB - 1) / TPB, TPB, 0, stream>>>(Wq, Wqb, nW);
  f32_to_bf16_kernel<<<(nW + TPB - 1) / TPB, TPB, 0, stream>>>(Wk, Wkb, nW);
  f32_to_bf16_kernel<<<(nW + TPB - 1) / TPB, TPB, 0, stream>>>(Wv, Wvb, nW);
  f32_to_bf16_kernel<<<(nW + TPB - 1) / TPB, TPB, 0, stream>>>(Wo, Wob, nW);

  const int nCache = BATCH * HEADS * 1024 * 128;  // 8,388,608
  cache_ingest_kernel<<<nCache / TPB, TPB, 0, stream>>>(cache, cacheOut, Kb, Vt);

  // 2048 waves (128 m-tiles x 16 n-tiles), 8 waves/block
  const int gemmBlocks = (MROWS / 32) * (DMODEL / 64) / 8;  // 256
  proj_gemm_kernel<MODE_Q><<<gemmBlocks, TPB, 0, stream>>>(Xq, Wqb, bq, Qb, nullptr);
  proj_gemm_kernel<MODE_K><<<gemmBlocks, TPB, 0, stream>>>(Xk, Wkb, bk, Kb, cacheOut);
  proj_gemm_kernel<MODE_V><<<gemmBlocks, TPB, 0, stream>>>(Xv, Wvb, bv, Vt, cacheOut);

  // 4096 waves (64 bh x 64 q-tiles), 8 waves/block
  attention_kernel<<<BATCH * HEADS * (LQ / 16) / 8, TPB, 0, stream>>>(Qb, Kb, Vt, Ao);

  proj_gemm_kernel<MODE_O><<<gemmBlocks, TPB, 0, stream>>>(Ao, Wob, bo, nullptr, out);

  (void)in_sizes; (void)n_in; (void)out_size; (void)ws_size;
}
